// XCA_51153060496106
// MI455X (gfx1250) — matrix operand order repaired and checked by reading, not compiled, not run on hardware
//
#include <hip/hip_runtime.h>
#include <math.h>

// ---------------------------------------------------------------------------
// XCA (cross-covariance attention) for MI455X / gfx1250, wave32 + WMMA f16.
// B=4 P=4 N=4096 E=512, HEADS=8, DH=64.
// Pipeline:
//   1) convert x, w_qkv, w_out to f16 in workspace
//   2) qkv = x @ w_qkv^T + b  (WMMA f32_16x16x32_f16, async double-buffered
//      LDS staging via global_load_async_to_lds_b128)
//   3) per-channel rsqrt norms of q,k over tokens
//   4) per (b,p,h): S = Q^T K (WMMA), scale by norms*temperature, softmax,
//      O = V @ P^T (WMMA), f16 output (b,p,n,512)
//   5) out = O @ w_out^T + b_out (WMMA, f32 output)
// ---------------------------------------------------------------------------

typedef __attribute__((ext_vector_type(16))) _Float16 v16h;
typedef __attribute__((ext_vector_type(8)))  _Float16 v8h;
typedef __attribute__((ext_vector_type(8)))  float    v8f;

union V16 { v16h v; v8h h[2]; };

#define B_      4
#define P_      4
#define N_      4096
#define E_      512
#define HEADS_  8
#define DH_     64
#define ROWS_   (B_ * P_ * N_)   // 65536 token rows

// CDNA5 async global->LDS copy, 16 bytes per lane, tracked by ASYNCcnt.
// VDST = LDS byte address (flat shared ptr truncates to LDS offset per ISA).
__device__ inline void async_copy_b128(void* lds, const void* gaddr) {
  asm volatile("global_load_async_to_lds_b128 %0, %1, off"
               :: "v"((unsigned int)(uintptr_t)lds), "v"(gaddr)
               : "memory");
}
__device__ inline void wait_async0() {
  asm volatile("s_wait_asynccnt 0x0" ::: "memory");
}

// ---------------------------------------------------------------------------
// f32 -> f16 elementwise convert (grid-stride)
// ---------------------------------------------------------------------------
__global__ void cvt_f32_f16(const float* __restrict__ src,
                            _Float16* __restrict__ dst, int n) {
  int i = blockIdx.x * blockDim.x + threadIdx.x;
  int stride = gridDim.x * blockDim.x;
  for (; i < n; i += stride) dst[i] = (_Float16)src[i];
}

// ---------------------------------------------------------------------------
// Tiled WMMA GEMM: C[M,N] = A[M,K] @ W[N,K]^T + bias[N]
// A, W are f16 row-major K-contiguous. 256 threads = 8 waves, block tile
// 128x128, each wave owns a 32x64 register tile (2x4 WMMA accumulators).
// Double-buffered LDS filled with async global->LDS b128 copies so the next
// K-tile streams in while the current tile's WMMAs execute.
// ---------------------------------------------------------------------------
template <bool HALF_OUT>
__global__ __launch_bounds__(256) void gemm_wmma(
    const _Float16* __restrict__ A, const _Float16* __restrict__ W,
    const float* __restrict__ bias, void* __restrict__ Cout,
    int M, int Ncols, int K) {
  __shared__ _Float16 shA[2][128][40];  // 32 K per step, +8 pad
  __shared__ _Float16 shB[2][128][40];

  const int t = threadIdx.x;
  const int lane = t & 31, wave = t >> 5;
  const int wrow = wave >> 1;        // 0..3 -> 32-row stripes
  const int wcol = wave & 1;         // 0..1 -> 64-col stripes
  const int m0 = blockIdx.x * 128;
  const int n0 = blockIdx.y * 128;

  v8f acc[2][4] = {};

  const int sr = t >> 1;             // staging row 0..127
  const int sh = (t & 1) * 16;       // staging K half: 0 or 16

  const _Float16* gA = A + (size_t)(m0 + sr) * K + sh;
  const _Float16* gW = W + (size_t)(n0 + sr) * K + sh;

  // prologue: stage K-step 0 into buffer 0
  async_copy_b128(&shA[0][sr][sh], gA);
  async_copy_b128(&shA[0][sr][sh + 8], gA + 8);
  async_copy_b128(&shB[0][sr][sh], gW);
  async_copy_b128(&shB[0][sr][sh + 8], gW + 8);

  int buf = 0;
  for (int k0 = 0; k0 < K; k0 += 32) {
    wait_async0();        // this wave's loads into `buf` have landed
    __syncthreads();      // ... and so have every other wave's

    if (k0 + 32 < K) {    // stream next K tile into the other buffer
      const int nb = buf ^ 1;
      async_copy_b128(&shA[nb][sr][sh],     gA + k0 + 32);
      async_copy_b128(&shA[nb][sr][sh + 8], gA + k0 + 40);
      async_copy_b128(&shB[nb][sr][sh],     gW + k0 + 32);
      async_copy_b128(&shB[nb][sr][sh + 8], gW + k0 + 40);
    }

    // A fragment: lane<16 -> row r, K0..7 | K16..23 ; lane>=16 -> K8..15 | K24..31
    const int fr = lane & 15;
    const int s  = (lane >> 4) * 8;
    const int kh = (lane >> 4) * 16;
    V16 afrag[2], bfrag[4];
    for (int mt = 0; mt < 2; mt++) {
      const _Float16* p = &shA[buf][wrow * 32 + mt * 16 + fr][0];
      afrag[mt].h[0] = *(const v8h*)(p + s);
      afrag[mt].h[1] = *(const v8h*)(p + 16 + s);
    }
    // B fragment: lane holds column n, K same order as A (s..s+7 | 16+s..16+s+7)
    for (int nt = 0; nt < 4; nt++) {
      const _Float16* p = &shB[buf][wcol * 64 + nt * 16 + fr][0];
      bfrag[nt].h[0] = *(const v8h*)(p + s);
      bfrag[nt].h[1] = *(const v8h*)(p + 16 + s);
    }
    for (int mt = 0; mt < 2; mt++)
      for (int nt = 0; nt < 4; nt++)
        acc[mt][nt] = __builtin_amdgcn_wmma_f32_16x16x32_f16(
            false, afrag[mt].v, false, bfrag[nt].v, (short)0, acc[mt][nt],
            false, false);
    buf ^= 1;
  }

  // C layout: VGPR i, lanes 0-15: row i, lanes 16-31: row i+8; col = lane&15
  const int fr = lane & 15;
  const int rh = (lane >> 4) * 8;
  for (int mt = 0; mt < 2; mt++)
    for (int nt = 0; nt < 4; nt++) {
      const int col = n0 + wcol * 64 + nt * 16 + fr;
      const float bv = bias[col];
      for (int i = 0; i < 8; i++) {
        const int row = m0 + wrow * 32 + mt * 16 + rh + i;
        const float v = acc[mt][nt][i] + bv;
        if (HALF_OUT)
          ((_Float16*)Cout)[(size_t)row * Ncols + col] = (_Float16)v;
        else
          ((float*)Cout)[(size_t)row * Ncols + col] = v;
      }
    }
}

// ---------------------------------------------------------------------------
// Per-channel 1/max(||.||,eps) over tokens for q,k (cols 0..1023 of qkv).
// Consecutive threads read consecutive channels -> coalesced.
// ---------------------------------------------------------------------------
__global__ __launch_bounds__(256) void rnorm_kernel(
    const _Float16* __restrict__ qkv, float* __restrict__ rsq) {
  const int bp  = blockIdx.x >> 2;
  const int col = ((blockIdx.x & 3) << 8) + threadIdx.x;  // 0..1023
  const _Float16* p = qkv + (size_t)bp * N_ * 1536 + col;
  float s = 0.f;
  for (int n = 0; n < N_; n++) {
    const float v = (float)p[(size_t)n * 1536];
    s += v * v;
  }
  rsq[bp * 1024 + col] = 1.f / fmaxf(sqrtf(s), 1e-12f);
}

// ---------------------------------------------------------------------------
// Fused channel attention per (b,p,h): 128 blocks x 128 threads (4 waves).
// Phase 1: S = Q^T K  (64x64, K=4096) -- q/k tiles transposed in LDS so both
//          WMMA fragments are contiguous vector LDS reads.
// Phase 2: logits scaled by rsq_q[d]*rsq_k[e]*temp[h], row softmax -> P (f16).
// Phase 3: O[n,d] = sum_e V[n,e] P[d,e]; P row-major in LDS is exactly the
//          B-fragment layout -> 8 fragments preloaded outside the token loop.
// ---------------------------------------------------------------------------
__global__ __launch_bounds__(128) void attn_kernel(
    const _Float16* __restrict__ qkv, const float* __restrict__ rsq,
    const float* __restrict__ temperature, _Float16* __restrict__ ob) {
  __shared__ _Float16 shQ[64][40];   // [channel d][token n] transposed
  __shared__ _Float16 shK[64][40];
  __shared__ float    shS[64][64];
  __shared__ _Float16 shP[64][72];

  const int bid = blockIdx.x;
  const int bp  = bid >> 3;
  const int h   = bid & 7;
  const int hc  = h * DH_;

  const int t = threadIdx.x;
  const int lane = t & 31, wave = t >> 5;  // 4 waves, wave = 16-row d stripe

  const _Float16* base = qkv + (size_t)bp * N_ * 1536;

  v8f acc[4] = {};

  const int sn = t >> 2;             // token within 32-block
  const int sc = (t & 3) * 16;       // channel base within 64

  // ---- Phase 1: S = Q^T K over K=4096 in steps of 32 ----
  for (int nb = 0; nb < N_ / 32; nb++) {
    __syncthreads();
    const int n0 = nb * 32;
    for (int j = 0; j < 16; j += 8) {  // coalesced load, transposed LDS store
      v8h q8 = *(const v8h*)(base + (size_t)(n0 + sn) * 1536 + hc + sc + j);
      v8h k8 = *(const v8h*)(base + (size_t)(n0 + sn) * 1536 + 512 + hc + sc + j);
      for (int i = 0; i < 8; i++) {
        shQ[sc + j + i][sn] = q8[i];
        shK[sc + j + i][sn] = k8[i];
      }
    }
    __syncthreads();

    const int fr = lane & 15;
    const int s  = (lane >> 4) * 8;
    const int kh = (lane >> 4) * 16;
    V16 a;
    const _Float16* pa = &shQ[wave * 16 + fr][0];
    a.h[0] = *(const v8h*)(pa + s);
    a.h[1] = *(const v8h*)(pa + 16 + s);
    for (int nt = 0; nt < 4; nt++) {
      V16 b;
      const _Float16* pb = &shK[nt * 16 + fr][0];
      b.h[0] = *(const v8h*)(pb + s);
      b.h[1] = *(const v8h*)(pb + 16 + s);
      acc[nt] = __builtin_amdgcn_wmma_f32_16x16x32_f16(
          false, a.v, false, b.v, (short)0, acc[nt], false, false);
    }
  }
  __syncthreads();
  {  // spill S stripes to LDS
    const int fr = lane & 15;
    const int rh = (lane >> 4) * 8;
    for (int nt = 0; nt < 4; nt++)
      for (int i = 0; i < 8; i++)
        shS[wave * 16 + rh + i][nt * 16 + fr] = acc[nt][i];
  }
  __syncthreads();

  // ---- Phase 2: fold norms + temperature, softmax over e ----
  if (t < 64) {
    const int d = t;
    const float rq = rsq[bp * 1024 + hc + d] * temperature[h];
    float vals[64];
    float m = -3.4e38f;
    for (int e = 0; e < 64; e++) {
      const float v = shS[d][e] * rq * rsq[bp * 1024 + 512 + hc + e];
      vals[e] = v;
      m = fmaxf(m, v);
    }
    float sum = 0.f;
    for (int e = 0; e < 64; e++) {
      const float p = __expf(vals[e] - m);
      vals[e] = p;
      sum += p;
    }
    const float inv = 1.f / sum;
    for (int e = 0; e < 64; e++) shP[d][e] = (_Float16)(vals[e] * inv);
  }
  __syncthreads();

  // ---- Phase 3: O[n, d] = sum_e V[n, e] * P[d, e] ----
  const int fr = lane & 15;
  const int s  = (lane >> 4) * 8;
  const int kh = (lane >> 4) * 16;
  V16 bfr[2][4];  // K=64 split in 2, N=64 split in 4; hoisted out of loop
  for (int kk = 0; kk < 2; kk++)
    for (int nt = 0; nt < 4; nt++) {
      const _Float16* pb = &shP[nt * 16 + fr][kk * 32];
      bfr[kk][nt].h[0] = *(const v8h*)(pb + s);
      bfr[kk][nt].h[1] = *(const v8h*)(pb + 16 + s);
    }

  const _Float16* vbase = base + 1024 + hc;  // V channels for this head
  for (int it = 0; it < N_ / 64; it++) {
    const int m0 = it * 64 + wave * 16;
    V16 a2[2];  // V rows straight from global: K-contiguous, 16B aligned
    for (int kk = 0; kk < 2; kk++) {
      const _Float16* pa = vbase + (size_t)(m0 + fr) * 1536 + kk * 32;
      a2[kk].h[0] = *(const v8h*)(pa + s);
      a2[kk].h[1] = *(const v8h*)(pa + 16 + s);
    }
    v8f oc[4] = {};
    for (int kk = 0; kk < 2; kk++)
      for (int nt = 0; nt < 4; nt++)
        oc[nt] = __builtin_amdgcn_wmma_f32_16x16x32_f16(
            false, a2[kk].v, false, bfr[kk][nt].v, (short)0, oc[nt], false,
            false);
    const int rh = (lane >> 4) * 8;
    for (int nt = 0; nt < 4; nt++)
      for (int i = 0; i < 8; i++)
        ob[((size_t)bp * N_ + m0 + rh + i) * 512 + hc + nt * 16 + fr] =
            (_Float16)oc[nt][i];
  }
}

// ---------------------------------------------------------------------------
extern "C" void kernel_launch(void* const* d_in, const int* in_sizes, int n_in,
                              void* d_out, int out_size, void* d_ws,
                              size_t ws_size, hipStream_t stream) {
  (void)in_sizes; (void)n_in; (void)out_size; (void)ws_size;
  const float* x    = (const float*)d_in[0];
  const float* wqkv = (const float*)d_in[1];
  const float* bqkv = (const float*)d_in[2];
  const float* temp = (const float*)d_in[3];
  const float* wout = (const float*)d_in[4];
  const float* bout = (const float*)d_in[5];

  char* ws = (char*)d_ws;
  size_t off = 0;
  auto alloc = [&](size_t bytes) -> void* {
    void* p = ws + off;
    off = (off + bytes + 255) & ~(size_t)255;
    return p;
  };

  _Float16* xb   = (_Float16*)alloc((size_t)ROWS_ * E_ * 2);     // 64 MB
  _Float16* wqb  = (_Float16*)alloc((size_t)1536 * 512 * 2);     // 1.5 MB
  _Float16* wob  = (_Float16*)alloc((size_t)512 * 512 * 2);      // 0.5 MB
  _Float16* qkvb = (_Float16*)alloc((size_t)ROWS_ * 1536 * 2);   // 192 MB
  _Float16* ab   = (_Float16*)alloc((size_t)ROWS_ * 512 * 2);    // 64 MB
  float*    rsq  = (float*)alloc((size_t)B_ * P_ * 1024 * 4);    // 64 KB

  // 1) f16 copies of x, weights
  cvt_f32_f16<<<2048, 256, 0, stream>>>(x, xb, ROWS_ * E_);
  cvt_f32_f16<<<768, 256, 0, stream>>>(wqkv, wqb, 1536 * 512);
  cvt_f32_f16<<<256, 256, 0, stream>>>(wout, wob, 512 * 512);

  // 2) qkv projection (f16 out, row stride 1536)
  dim3 g1(ROWS_ / 128, 1536 / 128);
  gemm_wmma<true><<<g1, 256, 0, stream>>>(xb, wqb, bqkv, qkvb, ROWS_, 1536, 512);

  // 3) q/k channel norms over tokens
  rnorm_kernel<<<B_ * P_ * 4, 256, 0, stream>>>(qkvb, rsq);

  // 4) fused channel attention per (b,p,h)
  attn_kernel<<<B_ * P_ * HEADS_, 128, 0, stream>>>(qkvb, rsq, temp, ab);

  // 5) output projection (f32 out)
  dim3 g2(ROWS_ / 128, 512 / 128);
  gemm_wmma<false><<<g2, 256, 0, stream>>>(ab, wob, bout, (float*)d_out,
                                           ROWS_, 512, 512);
}
